// MultiHeadAttention_2353642078756
// MI455X (gfx1250) — matrix operand order repaired and checked by reading, not compiled, not run on hardware
//
#include <hip/hip_runtime.h>

// ---------------------------------------------------------------------------
// MultiHeadAttention for MI455X (gfx1250, wave32, WMMA).
//   1. x -> bf16;  W_qkv -> bf16 transposed;  W_proj -> bf16 transposed
//   2. qkv = x @ W_qkv + b_qkv               (WMMA GEMM, bf16 out)
//   3. per (b,h) [batched over blockIdx.z]:
//        energy = Q @ K^T (f32) -> softmax/sqrt(768) -> bf16 -> @ V (bf16)
//   4. proj: out @ W_proj + b_proj -> d_out (f32)
// The torch "raw reshape" maps head-h row m to buffer row (m>>3) groups, so
// GEMMs use a grouped row-offset (m>>3)*ld1 + (m&7)*ld2 instead of copies.
// GEMM: BM=BN=128, BK=32, 8 waves of 64x32 (8 wmma / k-step / wave),
// single-buffered LDS; compiler software-pipelines the global loads.
// ---------------------------------------------------------------------------

typedef __bf16 bf16;
typedef __attribute__((ext_vector_type(16))) __bf16 v16bf;
typedef __attribute__((ext_vector_type(8)))  float  v8f;

#define BM 128
#define BN 128
#define BK 32
#define BKP 48   // padded LDS row (elements): 96B rows, 16B aligned

// ---------------------------------------------------------------------------
// C[M,N] = A[M,K] * Bt[N,K]^T (+ bias[N]);  z-batched via blockIdx.z strides.
// rowoff(m) = (m>>3)*ld1 + (m&7)*ld2 (elements); plain row-major: ld1=8L,ld2=L.
// Requires M%128==0, N%128==0, K%32==0, 16B-aligned bases/strides.
// ---------------------------------------------------------------------------
template <bool STORE_BF16, bool HAS_BIAS>
__global__ __launch_bounds__(256, 1) void gemm_wmma(
    const bf16* __restrict__ A, const bf16* __restrict__ Bt,
    void* __restrict__ Cp, const float* __restrict__ bias, int K,
    long long a1, long long a2, long long az,
    long long b1, long long b2, long long bz,
    long long c1, long long c2, long long cz)
{
    __shared__ bf16 sA[BM][BKP];
    __shared__ bf16 sB[BN][BKP];

    const int tid  = threadIdx.x;
    const int m0   = blockIdx.y * BM;
    const int n0   = blockIdx.x * BN;
    const long long z = blockIdx.z;
    A  += z * az;
    Bt += z * bz;
    const long long czo = z * cz;

    const int wid  = tid >> 5, lane = tid & 31;
    const int wm   = wid & 1,  wn   = wid >> 1;   // 2 x 4 waves; 64x32 per wave
    const int half = lane >> 4, l16 = lane & 15;

    v8f acc[4][2];
#pragma unroll
    for (int i = 0; i < 4; ++i)
#pragma unroll
        for (int j = 0; j < 2; ++j)
#pragma unroll
            for (int r = 0; r < 8; ++r) acc[i][j][r] = 0.0f;

    for (int k0 = 0; k0 < K; k0 += BK) {
        // --- stage A/B tiles (128x32 each): 2 x 16B per thread per matrix ---
#pragma unroll
        for (int s = 0; s < 2; ++s) {
            const int idx = tid + 256 * s;
            const int row = idx >> 2;            // 0..127
            const int kc  = (idx & 3) * 8;       // 0,8,16,24
            const int gm = m0 + row, gn = n0 + row;
            *reinterpret_cast<uint4*>(&sA[row][kc]) =
                *reinterpret_cast<const uint4*>(
                    A + (long long)(gm >> 3) * a1 + (long long)(gm & 7) * a2 + k0 + kc);
            *reinterpret_cast<uint4*>(&sB[row][kc]) =
                *reinterpret_cast<const uint4*>(
                    Bt + (long long)(gn >> 3) * b1 + (long long)(gn & 7) * b2 + k0 + kc);
        }
        __syncthreads();

        union Frag { uint4 q[2]; v16bf v; };
        Frag fa[4], fb[2];
        // A frag (ISA dense 16-bit A 16x32): lane<16 K={0..7,16..23}, lane>=16 K={8..15,24..31}
#pragma unroll
        for (int mt = 0; mt < 4; ++mt) {
            const int r = wm * 64 + mt * 16 + l16;
            fa[mt].q[0] = *reinterpret_cast<const uint4*>(&sA[r][8 * half]);
            fa[mt].q[1] = *reinterpret_cast<const uint4*>(&sA[r][16 + 8 * half]);
        }
        // B frag (column n per lane): lane<16 K={0..7,16..23}, lane>=16 K={8..15,24..31}
#pragma unroll
        for (int nt = 0; nt < 2; ++nt) {
            const int n = wn * 32 + nt * 16 + l16;
            fb[nt].q[0] = *reinterpret_cast<const uint4*>(&sB[n][8 * half]);
            fb[nt].q[1] = *reinterpret_cast<const uint4*>(&sB[n][16 + 8 * half]);
        }
#pragma unroll
        for (int mt = 0; mt < 4; ++mt)
#pragma unroll
            for (int nt = 0; nt < 2; ++nt)
                acc[mt][nt] = __builtin_amdgcn_wmma_f32_16x16x32_bf16(
                    false, fa[mt].v, false, fb[nt].v, (short)0, acc[mt][nt],
                    false, false);
        __syncthreads();
    }

    // ---- epilogue ----
#pragma unroll
    for (int mt = 0; mt < 4; ++mt) {
#pragma unroll
        for (int nt = 0; nt < 2; ++nt) {
            const int gn = n0 + wn * 32 + nt * 16 + l16;
            const float bv = HAS_BIAS ? bias[gn] : 0.0f;
#pragma unroll
            for (int r = 0; r < 8; ++r) {
                const int gm = m0 + wm * 64 + mt * 16 + half * 8 + r;
                const long long co =
                    czo + (long long)(gm >> 3) * c1 + (long long)(gm & 7) * c2 + gn;
                const float v = acc[mt][nt][r] + bv;
                if (STORE_BF16) reinterpret_cast<bf16*>(Cp)[co] = (bf16)v;
                else            reinterpret_cast<float*>(Cp)[co] = v;
            }
        }
    }
}

// f32 -> bf16 elementwise
__global__ __launch_bounds__(256) void convert_f32_bf16(
    const float* __restrict__ in, bf16* __restrict__ out, int n)
{
    for (int i = blockIdx.x * 256 + threadIdx.x; i < n; i += gridDim.x * 256)
        out[i] = (bf16)in[i];
}

// f32 [R][C] -> bf16 [C][R], LDS-tiled. R,C % 32 == 0.
__global__ __launch_bounds__(256) void transpose_f32_bf16(
    const float* __restrict__ in, bf16* __restrict__ out, int R, int C)
{
    __shared__ float t[32][33];
    const int c0 = blockIdx.x * 32, r0 = blockIdx.y * 32;
    const int tx = threadIdx.x, ty = threadIdx.y;   // (32,8)
#pragma unroll
    for (int j = ty; j < 32; j += 8)
        t[j][tx] = in[(long long)(r0 + j) * C + c0 + tx];
    __syncthreads();
#pragma unroll
    for (int j = ty; j < 32; j += 8)
        out[(long long)(c0 + j) * R + r0 + tx] = (bf16)t[tx][j];
}

// V (grouped layout inside qkv buffer) -> Vt[z][768][1024] bf16, z=(b*8+h)
__global__ __launch_bounds__(256) void vt_transpose(
    const bf16* __restrict__ qkvb, bf16* __restrict__ vt)
{
    __shared__ bf16 t[32][33];
    const int z = blockIdx.z;
    const bf16* src = qkvb + (long long)(128 * z) * 18432 + 2 * 6144;
    bf16* dst = vt + (long long)z * 768 * 1024;
    const int m0 = blockIdx.x * 32, d0 = blockIdx.y * 32;
    const int tx = threadIdx.x, ty = threadIdx.y;   // (32,8)
#pragma unroll
    for (int j = ty; j < 32; j += 8) {
        const int m = m0 + j;
        t[j][tx] = src[(long long)(m >> 3) * 18432 + (m & 7) * 768 + d0 + tx];
    }
    __syncthreads();
#pragma unroll
    for (int j = ty; j < 32; j += 8)
        dst[(long long)(d0 + j) * 1024 + m0 + tx] = t[tx][j];
}

// softmax over a 1024 row then * post (=1/sqrt(768)); z = blockIdx.y slice
__global__ __launch_bounds__(256) void softmax_row(
    const float* __restrict__ E, bf16* __restrict__ P, float post)
{
    __shared__ float red[256];
    const int row = blockIdx.x, tid = threadIdx.x;
    const long long zoff = (long long)blockIdx.y << 20;   // 1024*1024
    const float* e = E + zoff + (long long)row * 1024;
    float x[4];
#pragma unroll
    for (int i = 0; i < 4; ++i) x[i] = e[tid + 256 * i];
    float m = fmaxf(fmaxf(x[0], x[1]), fmaxf(x[2], x[3]));
    red[tid] = m; __syncthreads();
    for (int s = 128; s > 0; s >>= 1) {
        if (tid < s) red[tid] = fmaxf(red[tid], red[tid + s]);
        __syncthreads();
    }
    m = red[0]; __syncthreads();
    float sum = 0.0f;
#pragma unroll
    for (int i = 0; i < 4; ++i) { x[i] = __expf(x[i] - m); sum += x[i]; }
    red[tid] = sum; __syncthreads();
    for (int s = 128; s > 0; s >>= 1) {
        if (tid < s) red[tid] += red[tid + s];
        __syncthreads();
    }
    const float inv = post / red[0];
    bf16* p = P + zoff + (long long)row * 1024;
#pragma unroll
    for (int i = 0; i < 4; ++i) p[tid + 256 * i] = (bf16)(x[i] * inv);
}

extern "C" void kernel_launch(void* const* d_in, const int* in_sizes, int n_in,
                              void* d_out, int out_size, void* d_ws, size_t ws_size,
                              hipStream_t stream)
{
    (void)in_sizes; (void)n_in; (void)out_size;
    const float* x     = (const float*)d_in[0];   // [2,1024,768]
    const float* Wqkv  = (const float*)d_in[1];   // [768,18432]
    const float* bqkv  = (const float*)d_in[2];   // [18432]
    const float* Wproj = (const float*)d_in[3];   // [6144,768]
    const float* bproj = (const float*)d_in[4];   // [768]
    float* out = (float*)d_out;                   // [2,1024,768]

    // full-batch path needs ~267 MB; fallback per-head loop needs ~173 MB
    const bool fullz = ws_size >= 267386880ULL;

    char* ws = (char*)d_ws;
    size_t off = 0;
    auto take = [&](size_t bytes) {
        char* p = ws + off;
        off = (off + bytes + 255) & ~(size_t)255;
        return p;
    };
    bf16* xb     = (bf16*)take(2048ULL * 768 * 2);
    bf16* wqkvT  = (bf16*)take(18432ULL * 768 * 2);
    bf16* wprojT = (bf16*)take(768ULL * 6144 * 2);
    bf16* qkvb   = (bf16*)take(2048ULL * 18432 * 2);
    bf16* vtb    = (bf16*)take(16ULL * 768 * 1024 * 2);
    bf16* outb   = (bf16*)take(2048ULL * 6144 * 2);
    const size_t eslices = fullz ? 16 : 1;
    float* energy = (float*)take(eslices * 1024ULL * 1024 * 4);
    bf16*  attnb  = (bf16*)take(eslices * 1024ULL * 1024 * 2);

    // 1. precision conversion + weight transposes
    convert_f32_bf16<<<6144, 256, 0, stream>>>(x, xb, 2048 * 768);
    transpose_f32_bf16<<<dim3(576, 24), dim3(32, 8), 0, stream>>>(Wqkv, wqkvT, 768, 18432);
    transpose_f32_bf16<<<dim3(24, 192), dim3(32, 8), 0, stream>>>(Wproj, wprojT, 6144, 768);

    // 2. qkv = x @ W_qkv + b_qkv   (M=2048, N=18432, K=768)
    gemm_wmma<true, true><<<dim3(144, 16), 256, 0, stream>>>(
        xb, wqkvT, qkvb, bqkv, 768,
        8LL * 768, 768, 0, 8LL * 768, 768, 0, 8LL * 18432, 18432, 0);

    // 3. V^T for all 16 (b,h)
    vt_transpose<<<dim3(32, 24, 16), dim3(32, 8), 0, stream>>>(qkvb, vtb);

    // 4. attention; z = b*8 + h, head base row = 128*z (raw reshape)
    if (fullz) {
        gemm_wmma<false, false><<<dim3(8, 8, 16), 256, 0, stream>>>(
            qkvb, qkvb + 6144, energy, nullptr, 768,
            18432, 768, 128LL * 18432, 18432, 768, 128LL * 18432,
            8LL * 1024, 1024, 1024LL * 1024);
        softmax_row<<<dim3(1024, 16), 256, 0, stream>>>(
            energy, attnb, 0.03608439182435161f);
        gemm_wmma<true, false><<<dim3(6, 8, 16), 256, 0, stream>>>(
            attnb, vtb, outb, nullptr, 1024,
            8LL * 1024, 1024, 1024LL * 1024, 8LL * 1024, 1024, 768LL * 1024,
            6144, 768, 128LL * 6144);
    } else {
        for (int z = 0; z < 16; ++z) {
            const bf16* Q = qkvb + (long long)(128 * z) * 18432;
            gemm_wmma<false, false><<<dim3(8, 8), 256, 0, stream>>>(
                Q, Q + 6144, energy, nullptr, 768,
                18432, 768, 0, 18432, 768, 0, 8LL * 1024, 1024, 0);
            softmax_row<<<dim3(1024, 1), 256, 0, stream>>>(
                energy, attnb, 0.03608439182435161f);
            gemm_wmma<true, false><<<dim3(6, 8), 256, 0, stream>>>(
                attnb, vtb + (long long)z * 768 * 1024,
                outb + (long long)(128 * z) * 6144, nullptr, 1024,
                8LL * 1024, 1024, 0, 8LL * 1024, 1024, 0, 6144, 768, 0);
        }
    }

    // 5. projection: d_out = outb @ W_proj + b_proj (M=2048, N=768, K=6144)
    gemm_wmma<false, true><<<dim3(6, 16), 256, 0, stream>>>(
        outb, wprojT, out, bproj, 6144,
        8LL * 6144, 6144, 0, 8LL * 6144, 6144, 0, 8LL * 768, 768, 0);
}
